// MultiHeadAttention_31619549233752
// MI455X (gfx1250) — matrix operand order repaired and checked by reading, not compiled, not run on hardware
//
#include <hip/hip_runtime.h>
#include <hip/hip_bf16.h>

typedef _Float16 h16;
typedef __attribute__((ext_vector_type(16))) _Float16 v16h;
typedef __attribute__((ext_vector_type(8)))  _Float16 v8h;
typedef __attribute__((ext_vector_type(8)))  float    v8f;

#define BB   2048
#define LL   64
#define DDIM 256
#define HH   8
#define DKK  32
#define DPP  12
#define DFFN 1024
#define BLR  (BB * LL)          // 131072 rows
#define INV_TEMP 0.17677669529663687f  // 1/sqrt(32)

// LDS byte offset of a __shared__ pointer (flat LDS aperture: addr[31:0] = LDS addr)
#define LDS_OFF(p) ((unsigned)(uintptr_t)(p))

// Per-lane async 16B global -> LDS copy (ASYNCcnt tracked, no VGPR round-trip)
__device__ __forceinline__ void async_b128(unsigned lds_off, const void* gaddr) {
  asm volatile("global_load_async_to_lds_b128 %0, %1, off"
               :: "v"(lds_off), "v"(gaddr) : "memory");
}
__device__ __forceinline__ void wait_async0() {
  asm volatile("s_wait_asynccnt 0" ::: "memory");
}

// ---------------------------------------------------------------------------
// LayerNorm over rows of 256, output f16. One wave (32 lanes) per row.
// ---------------------------------------------------------------------------
__global__ __launch_bounds__(256) void ln_kernel(const float* __restrict__ x,
                                                 const float* __restrict__ g,
                                                 const float* __restrict__ b,
                                                 h16* __restrict__ out) {
  int wave = threadIdx.x >> 5;
  int lane = threadIdx.x & 31;
  long row = (long)blockIdx.x * 8 + wave;
  const float* xr = x + row * DDIM;

  float vals[8];
  float s = 0.f;
#pragma unroll
  for (int i = 0; i < 8; ++i) { vals[i] = xr[lane + 32 * i]; s += vals[i]; }
#pragma unroll
  for (int off = 16; off > 0; off >>= 1) s += __shfl_xor(s, off, 32);
  float mu = s * (1.f / DDIM);

  float vs = 0.f;
#pragma unroll
  for (int i = 0; i < 8; ++i) { float d = vals[i] - mu; vs += d * d; }
#pragma unroll
  for (int off = 16; off > 0; off >>= 1) vs += __shfl_xor(vs, off, 32);
  float rstd = rsqrtf(vs * (1.f / DDIM) + 1e-6f);

#pragma unroll
  for (int i = 0; i < 8; ++i) {
    int c = lane + 32 * i;
    out[row * DDIM + c] = (h16)((vals[i] - mu) * rstd * g[c] + b[c]);
  }
}

// ---------------------------------------------------------------------------
// f32 -> f16 conversion (grid-stride)
// ---------------------------------------------------------------------------
__global__ void convert_kernel(const float* __restrict__ src,
                               h16* __restrict__ dst, long n) {
  long i = (long)blockIdx.x * blockDim.x + threadIdx.x;
  long stride = (long)gridDim.x * blockDim.x;
  for (; i < n; i += stride) dst[i] = (h16)src[i];
}

// ---------------------------------------------------------------------------
// f32 [K,N] -> f16 transposed [N,K] (weights; done once, read ~1024x each)
// ---------------------------------------------------------------------------
__global__ void convert_transpose_kernel(const float* __restrict__ src,
                                         h16* __restrict__ dst, int K, int N) {
  long n = (long)K * N;
  long i = (long)blockIdx.x * blockDim.x + threadIdx.x;
  long stride = (long)gridDim.x * blockDim.x;
  for (; i < n; i += stride) {
    int kk = (int)(i / N), nn = (int)(i % N);
    dst[(long)nn * K + kk] = (h16)src[i];
  }
}

// ---------------------------------------------------------------------------
// WMMA GEMM: C[M,N] = A[M,K](f16) * B[K,N] with B pre-transposed as BT[N,K].
// Block: 128 threads / 4 waves; 128x64 tile; wave w owns rows 32w..32w+31
// (8 v_wmma_f32_16x16x32_f16 per K-step). Tiles staged with async b128 copies.
// ---------------------------------------------------------------------------
__global__ __launch_bounds__(128) void gemm_kernel(
    const h16* __restrict__ A, const h16* __restrict__ BT,
    int M, int N, int K,
    h16* __restrict__ out16, float* __restrict__ out32,
    const float* __restrict__ bias, const float* __restrict__ resid,
    int relu) {
  __shared__ h16 As[128][40];  // 128 rows x 32 K (pad 40: conflict-free, 16B aligned)
  __shared__ h16 Bs[64][40];   // Bs[n][k] = B[k][n0+n]

  const int ntiles = N >> 6;
  const int mtile = blockIdx.x / ntiles;
  const int ntile = blockIdx.x % ntiles;
  const long m0 = (long)mtile * 128;
  const int n0 = ntile * 64;

  const int tid = threadIdx.x;
  const int wave = tid >> 5;
  const int lane = tid & 31;
  const int half = lane >> 4;
  const int r16 = lane & 15;

  v8f acc[2][4];
#pragma unroll
  for (int s = 0; s < 2; ++s)
#pragma unroll
    for (int nt = 0; nt < 4; ++nt)
#pragma unroll
      for (int i = 0; i < 8; ++i) acc[s][nt][i] = 0.f;

  // Stagers: A row `tid` (64B), BT row `tid>>1` half `(tid&1)*16` (32B)
  const unsigned lA = LDS_OFF(&As[tid][0]);
  const h16* gA = A + (m0 + tid) * (long)K;
  const unsigned lB = LDS_OFF(&Bs[tid >> 1][(tid & 1) * 16]);
  const h16* gB = BT + (long)(n0 + (tid >> 1)) * K + (tid & 1) * 16;

  for (int k0 = 0; k0 < K; k0 += 32) {
#pragma unroll
    for (int j = 0; j < 4; ++j) async_b128(lA + j * 16, gA + k0 + j * 8);
#pragma unroll
    for (int j = 0; j < 2; ++j) async_b128(lB + j * 16, gB + k0 + j * 8);
    if (k0 + 32 < K) __builtin_prefetch(gA + k0 + 32, 0, 1);
    wait_async0();
    __syncthreads();

#pragma unroll
    for (int s = 0; s < 2; ++s) {
      // A fragment: rows wave*32 + s*16 + r16; half picks K 0-7/16-23 vs 8-15/24-31
      v16h a;
      {
        int row = wave * 32 + s * 16 + r16;
        v8h lo = *(const v8h*)&As[row][half * 8];
        v8h hi = *(const v8h*)&As[row][16 + half * 8];
#pragma unroll
        for (int i = 0; i < 8; ++i) { a[i] = lo[i]; a[8 + i] = hi[i]; }
      }
#pragma unroll
      for (int nt = 0; nt < 4; ++nt) {
        // B fragment: half picks K 0-7/16-23 vs 8-15/24-31 (same as A), from Bs[col][k]
        v16h bf;
        int col = nt * 16 + r16;
        v8h lo = *(const v8h*)&Bs[col][half * 8];
        v8h hi = *(const v8h*)&Bs[col][16 + half * 8];
#pragma unroll
        for (int i = 0; i < 8; ++i) { bf[i] = lo[i]; bf[8 + i] = hi[i]; }
        acc[s][nt] = __builtin_amdgcn_wmma_f32_16x16x32_f16(
            false, a, false, bf, (short)0, acc[s][nt], false, false);
      }
    }
    __syncthreads();
  }

  // epilogue: D element c[r] -> row = +half*8+r, col = nt*16 + r16
#pragma unroll
  for (int s = 0; s < 2; ++s) {
#pragma unroll
    for (int nt = 0; nt < 4; ++nt) {
#pragma unroll
      for (int r = 0; r < 8; ++r) {
        int row = wave * 32 + s * 16 + half * 8 + r;
        int col = n0 + nt * 16 + r16;
        float v = acc[s][nt][r];
        if (bias) v += bias[col];
        if (relu) v = v > 0.f ? v : 0.f;
        long idx = (m0 + row) * (long)N + col;
        if (resid) v += resid[idx];
        if (out32) out32[idx] = v;
        if (out16) out16[idx] = (h16)v;
      }
    }
  }
}

// ---------------------------------------------------------------------------
// qeff = (qh + pose @ Wp) * (1/sqrt(DK)), in place on the f16 qh buffer.
// ---------------------------------------------------------------------------
__global__ __launch_bounds__(256) void pose_bias_kernel(
    const float* __restrict__ pose, const float* __restrict__ Wp,
    h16* __restrict__ qeff) {
  long row = blockIdx.x;
  int n = threadIdx.x;
  float s = 0.f;
#pragma unroll
  for (int p = 0; p < DPP; ++p) s += pose[row * DPP + p] * Wp[p * DDIM + n];
  long idx = row * DDIM + n;
  float qv = (float)qeff[idx] + s;
  qeff[idx] = (h16)(qv * INV_TEMP);
}

// ---------------------------------------------------------------------------
// Attention per (b,h): logits = Qeff·K^T (WMMA), mask+softmax (attn -> d_out),
// out = attn·V (WMMA), x2 = out + residual(q).
// ---------------------------------------------------------------------------
__global__ __launch_bounds__(128) void attn_kernel(
    const h16* __restrict__ qeff, const h16* __restrict__ kh,
    const h16* __restrict__ vh, const int* __restrict__ mask,
    const float* __restrict__ qin, float* __restrict__ attn_out,
    float* __restrict__ x2) {
  __shared__ h16 Qs[64][40];
  __shared__ h16 Ks[64][40];
  __shared__ h16 Vt[32][72];     // Vt[d][j] = V[j][d]
  __shared__ float Ls[64][66];   // logits
  __shared__ h16 Ps[64][72];     // softmax probs (f16) for second GEMM

  const int bh = blockIdx.x;
  const int b = bh >> 3, h = bh & 7;
  const int tid = threadIdx.x;
  const int wave = tid >> 5, lane = tid & 31;
  const int half = lane >> 4, r16 = lane & 15;
  const long base = ((long)b * LL) * DDIM + h * DKK;

  {  // stage Q', K (async b128) and V^T (scalar transpose): row l = tid>>1
    int l = tid >> 1, c0 = (tid & 1) * 16;
    const h16* sq = qeff + base + (long)l * DDIM + c0;
    const h16* sk = kh + base + (long)l * DDIM + c0;
    unsigned lq = LDS_OFF(&Qs[l][c0]);
    unsigned lk = LDS_OFF(&Ks[l][c0]);
#pragma unroll
    for (int j = 0; j < 2; ++j) {
      async_b128(lq + j * 16, sq + j * 8);
      async_b128(lk + j * 16, sk + j * 8);
    }
    const h16* sv = vh + base + (long)l * DDIM + c0;
#pragma unroll
    for (int e = 0; e < 16; ++e) Vt[c0 + e][l] = sv[e];
    wait_async0();
  }
  __syncthreads();

  // ---- logits: 16x64 strip per wave, single K-step of 32 ----
  const int row = wave * 16 + r16;
  v16h a;
  {
    v8h lo = *(const v8h*)&Qs[row][half * 8];
    v8h hi = *(const v8h*)&Qs[row][16 + half * 8];
#pragma unroll
    for (int i = 0; i < 8; ++i) { a[i] = lo[i]; a[8 + i] = hi[i]; }
  }
#pragma unroll
  for (int nt = 0; nt < 4; ++nt) {
    v16h bf;  // bf: K[nt*16+r16][half*8+e] (e<8), [16+half*8+e-8] (e>=8)
    int col = nt * 16 + r16;
    v8h lo = *(const v8h*)&Ks[col][half * 8];
    v8h hi = *(const v8h*)&Ks[col][16 + half * 8];
#pragma unroll
    for (int i = 0; i < 8; ++i) { bf[i] = lo[i]; bf[8 + i] = hi[i]; }
    v8f c;
#pragma unroll
    for (int i = 0; i < 8; ++i) c[i] = 0.f;
    c = __builtin_amdgcn_wmma_f32_16x16x32_f16(false, a, false, bf, (short)0,
                                               c, false, false);
#pragma unroll
    for (int r = 0; r < 8; ++r)
      Ls[wave * 16 + half * 8 + r][nt * 16 + r16] = c[r];
  }
  __syncthreads();

  // ---- masked softmax: 2 threads per row, 32 cols each, pair via shfl ----
  {
    int qrow = tid >> 1, seg = tid & 1;
    const int* mrow = mask + ((long)b * LL + qrow) * LL + seg * 32;
    float lv[32];
    float mx = -3.4e38f;
#pragma unroll
    for (int j = 0; j < 32; ++j) {
      float t = (mrow[j] == 0) ? -1e9f : Ls[qrow][seg * 32 + j];
      lv[j] = t;
      mx = fmaxf(mx, t);
    }
    mx = fmaxf(mx, __shfl_xor(mx, 1, 32));
    float s = 0.f;
#pragma unroll
    for (int j = 0; j < 32; ++j) { lv[j] = expf(lv[j] - mx); s += lv[j]; }
    s += __shfl_xor(s, 1, 32);
    float inv = 1.f / s;
    float* arow = attn_out + ((long)bh * LL + qrow) * LL + seg * 32;
#pragma unroll
    for (int j = 0; j < 32; ++j) {
      float p = lv[j] * inv;
      arow[j] = p;
      Ps[qrow][seg * 32 + j] = (h16)p;
    }
  }
  __syncthreads();

  // ---- out = P(64x64) @ V(64x32): 2 K-steps of 32, 2 N-tiles of 16 ----
  v8f oc[2];
#pragma unroll
  for (int nt = 0; nt < 2; ++nt)
#pragma unroll
    for (int i = 0; i < 8; ++i) oc[nt][i] = 0.f;

#pragma unroll
  for (int ks = 0; ks < 2; ++ks) {
    v16h ap;  // A frag of P at K offset ks*32
    {
      v8h lo = *(const v8h*)&Ps[row][ks * 32 + half * 8];
      v8h hi = *(const v8h*)&Ps[row][ks * 32 + 16 + half * 8];
#pragma unroll
      for (int i = 0; i < 8; ++i) { ap[i] = lo[i]; ap[8 + i] = hi[i]; }
    }
#pragma unroll
    for (int nt = 0; nt < 2; ++nt) {
      v16h bf;  // bf: V[ks*32 + half*8 + e] (e<8), V[ks*32 + 16 + half*8 + e-8] (e>=8), col nt*16+r16
      int col = nt * 16 + r16;
      v8h lo = *(const v8h*)&Vt[col][ks * 32 + half * 8];
      v8h hi = *(const v8h*)&Vt[col][ks * 32 + 16 + half * 8];
#pragma unroll
      for (int i = 0; i < 8; ++i) { bf[i] = lo[i]; bf[8 + i] = hi[i]; }
      oc[nt] = __builtin_amdgcn_wmma_f32_16x16x32_f16(
          false, ap, false, bf, (short)0, oc[nt], false, false);
    }
  }

  // x2 = out + residual(q)
#pragma unroll
  for (int nt = 0; nt < 2; ++nt) {
#pragma unroll
    for (int r = 0; r < 8; ++r) {
      int orow = wave * 16 + half * 8 + r;
      int ocol = nt * 16 + r16;
      long idx = base + (long)orow * DDIM + ocol;
      x2[idx] = oc[nt][r] + qin[idx];
    }
  }
}

// ---------------------------------------------------------------------------
extern "C" void kernel_launch(void* const* d_in, const int* in_sizes, int n_in,
                              void* d_out, int out_size, void* d_ws,
                              size_t ws_size, hipStream_t stream) {
  (void)in_sizes; (void)n_in; (void)out_size; (void)ws_size;

  const float* q    = (const float*)d_in[0];
  const float* k    = (const float*)d_in[1];
  const float* v    = (const float*)d_in[2];
  const int*   mask = (const int*)d_in[3];
  const float* pose = (const float*)d_in[4];
  const float* Wq   = (const float*)d_in[5];
  const float* Wk   = (const float*)d_in[6];
  const float* Wv   = (const float*)d_in[7];
  const float* Wp   = (const float*)d_in[8];
  const float* fc1w = (const float*)d_in[9];
  const float* fc1b = (const float*)d_in[10];
  const float* fc2w = (const float*)d_in[11];
  const float* fc2b = (const float*)d_in[12];
  const float* lag  = (const float*)d_in[13];
  const float* lab  = (const float*)d_in[14];
  const float* lfg  = (const float*)d_in[15];
  const float* lfb  = (const float*)d_in[16];

  float* out_y    = (float*)d_out;                         // [B,L,D]
  float* out_attn = out_y + (size_t)BLR * DDIM;            // [B,H,L,L]
  float* out_v    = out_attn + (size_t)BB * HH * LL * LL;  // [B,L,D]

  char* w = (char*)d_ws;
  const size_t b16 = (size_t)BLR * DDIM * sizeof(h16);
  h16*   xln  = (h16*)(w);                 // reused later as yln
  h16*   k16  = (h16*)(w + 1 * b16);       // reused (with v16,qeff,kh16) as h1
  h16*   v16  = (h16*)(w + 2 * b16);
  h16*   qeff = (h16*)(w + 3 * b16);
  h16*   kh16 = (h16*)(w + 4 * b16);
  h16*   vh16 = (h16*)(w + 5 * b16);
  float* x2   = (float*)(w + 6 * b16);     // [BLR, D] f32
  char* wb = w + 6 * b16 + (size_t)BLR * DDIM * sizeof(float);
  h16* wq16 = (h16*)wb;  wb += (size_t)DDIM * DDIM * 2;    // all stored [N,K]
  h16* wk16 = (h16*)wb;  wb += (size_t)DDIM * DDIM * 2;
  h16* wv16 = (h16*)wb;  wb += (size_t)DDIM * DDIM * 2;
  h16* f1w16 = (h16*)wb; wb += (size_t)DDIM * DFFN * 2;
  h16* f2w16 = (h16*)wb;
  h16* h1  = k16;   // 256 MiB overlay across k16..kh16 (dead after attention)
  h16* yln = xln;

  // 1) LN(q) -> f16; convert k, v; weights -> f16 transposed [N,K]
  ln_kernel<<<BLR / 8, 256, 0, stream>>>(q, lag, lab, xln);
  convert_kernel<<<2048, 256, 0, stream>>>(k, k16, (long)BLR * DDIM);
  convert_kernel<<<2048, 256, 0, stream>>>(v, v16, (long)BLR * DDIM);
  convert_transpose_kernel<<<64, 256, 0, stream>>>(Wq, wq16, DDIM, DDIM);
  convert_transpose_kernel<<<64, 256, 0, stream>>>(Wk, wk16, DDIM, DDIM);
  convert_transpose_kernel<<<64, 256, 0, stream>>>(Wv, wv16, DDIM, DDIM);
  convert_transpose_kernel<<<256, 256, 0, stream>>>(fc1w, f1w16, DDIM, DFFN);
  convert_transpose_kernel<<<256, 256, 0, stream>>>(fc2w, f2w16, DFFN, DDIM);

  // 2) Projections (WMMA GEMMs, M=131072, K=256, N=256)
  dim3 gproj((BLR / 128) * (DDIM / 64));
  gemm_kernel<<<gproj, 128, 0, stream>>>(xln, wq16, BLR, DDIM, DDIM,
                                         qeff, nullptr, nullptr, nullptr, 0);
  gemm_kernel<<<gproj, 128, 0, stream>>>(k16, wk16, BLR, DDIM, DDIM,
                                         kh16, nullptr, nullptr, nullptr, 0);
  gemm_kernel<<<gproj, 128, 0, stream>>>(v16, wv16, BLR, DDIM, DDIM,
                                         vh16, out_v, nullptr, nullptr, 0);

  // 3) qeff = (qh + pose@Wp) / sqrt(DK)
  pose_bias_kernel<<<BLR, 256, 0, stream>>>(pose, Wp, qeff);

  // 4) Attention (WMMA) + residual -> x2, attn probs -> d_out
  attn_kernel<<<BB * HH, 128, 0, stream>>>(qeff, kh16, vh16, mask, q,
                                           out_attn, x2);

  // 5) FFN: LN(x2) -> f16; fc1 (relu) -> h1; fc2 + bias + residual -> out_y
  ln_kernel<<<BLR / 8, 256, 0, stream>>>(x2, lfg, lfb, yln);
  gemm_kernel<<<(BLR / 128) * (DFFN / 64), 128, 0, stream>>>(
      yln, f1w16, BLR, DFFN, DDIM, h1, nullptr, fc1b, nullptr, 1);
  gemm_kernel<<<(BLR / 128) * (DDIM / 64), 128, 0, stream>>>(
      h1, f2w16, BLR, DDIM, DFFN, nullptr, out_y, fc2b, x2, 0);
}
